// DSWNV_84387517432212
// MI455X (gfx1250) — matrix operand order repaired and checked by reading, not compiled, not run on hardware
//
#include <hip/hip_runtime.h>

typedef __attribute__((ext_vector_type(16))) _Float16 v16h;
typedef __attribute__((ext_vector_type(8)))  float    v8f;

#define BDIM 256

constexpr int B_    = 4;
constexpr int T_    = 8249;
constexpr int TAUX  = 75;
constexpr int UPF   = 110;
constexpr int NAUX  = 54;
constexpr int NQ    = 256;
constexpr int HID   = 192;
constexpr int SKIPC = 256;
constexpr int KW    = 6;
constexpr int PADH  = 180;                 // max left pad = 36*5
constexpr int HSTR  = PADH + T_ + 16;      // 8445 (16 zero slack on right)
constexpr int ASTR  = 5 + T_ + 16;         // 8270
constexpr int NT16  = (T_ + 15) / 16;      // 516

// ---------------------------------------------------------------- WMMA utils
__device__ __forceinline__ v8f wmma16(v16h a, v16h b, v8f c) {
  return __builtin_amdgcn_wmma_f32_16x16x32_f16(false, a, false, b, (short)0, c,
                                                false, false);
}

// A operand: 16x32 (MxK) f16, row stride = rs halves. Per ISA 7.12.2:
// lanes 0-15 hold M=lane, K {0..7,16..23}; lanes 16-31 hold M=lane-16, K {8..15,24..31}
__device__ __forceinline__ v16h loadA(const _Float16* W, int rs, int lane) {
  int m = lane & 15, hi = lane >> 4;
  const _Float16* p = W + m * rs;
  v16h a;
#pragma unroll
  for (int i = 0; i < 8; ++i) a[i] = p[hi * 8 + i];
#pragma unroll
  for (int i = 0; i < 8; ++i) a[8 + i] = p[16 + hi * 8 + i];
  return a;
}

// B operand: 32x16 (KxN) f16 from [n][c] layout (n stride = ns halves).
// lanes 0-15: N=lane, K {0..7,16..23}; lanes 16-31: N=lane-16, K {8..15,24..31}
__device__ __forceinline__ v16h loadB(const _Float16* Bp, int ns, int lane) {
  int n = lane & 15, hi = lane >> 4;
  const _Float16* p = Bp + n * ns;
  v16h b;
#pragma unroll
  for (int i = 0; i < 16; ++i) b[i] = p[(i >> 3) * 16 + hi * 8 + (i & 7)];
  return b;
}

// ---------------------------------------------------------------- aux chain
__global__ void k_aux_scale(const float* aux, const float* w, const float* bia,
                            float* a0) {
  int i = blockIdx.x * blockDim.x + threadIdx.x;
  if (i >= B_ * NAUX * TAUX) return;
  int t = i % TAUX, c = (i / TAUX) % NAUX, b = i / (TAUX * NAUX);
  float s = bia[c];
  for (int ci = 0; ci < NAUX; ++ci)
    s += w[c * NAUX + ci] * aux[(b * NAUX + ci) * TAUX + t];
  a0[i] = s;
}

__global__ void k_aux0(const float* a0, const float* w, const float* bia,
                       float* a1) {
  int i = blockIdx.x * blockDim.x + threadIdx.x;
  if (i >= B_ * 162 * TAUX) return;
  int t = i % TAUX, o = (i / TAUX) % 162, b = i / (TAUX * 162);
  float s = bia[o];
  for (int k = 0; k < 3; ++k) {
    int tt = t + k - 1;
    if (tt < 0 || tt >= TAUX) continue;
    for (int ci = 0; ci < NAUX; ++ci)
      s += w[(o * NAUX + ci) * 3 + k] * a0[(b * NAUX + ci) * TAUX + tt];
  }
  a1[i] = s;
}

__global__ void k_aux1(const float* a1, const float* w, const float* bia,
                       float* a2) {
  int i = blockIdx.x * blockDim.x + threadIdx.x;
  if (i >= B_ * 486 * TAUX) return;
  int t = i % TAUX, o = (i / TAUX) % 486, b = i / (TAUX * 486);
  float s = bia[o];
  for (int k = 0; k < 3; ++k) {
    int tt = t + 3 * k - 3;
    if (tt < 0 || tt >= TAUX) continue;
    for (int ci = 0; ci < 162; ++ci)
      s += w[(o * 162 + ci) * 3 + k] * a1[(b * 162 + ci) * TAUX + tt];
  }
  a2[i] = s;
}

// g[l,b,j,q] = sum_c inx_w[l,j,c] * a2[b,c,q]
__global__ void k_cond(const float* a2, const float* inxw, float* g) {
  int i = blockIdx.x * blockDim.x + threadIdx.x;
  if (i >= 9 * B_ * 384 * TAUX) return;
  int q = i % TAUX, j = (i / TAUX) % 384, b = (i / (TAUX * 384)) % B_,
      l = i / (TAUX * 384 * B_);
  float s = 0.f;
  const float* wp = inxw + (l * 384 + j) * 486;
  for (int c = 0; c < 486; ++c) s += wp[c] * a2[(b * 486 + c) * TAUX + q];
  g[((l * B_ + b) * 384 + j) * TAUX + q] = s;
}

// wsum[l,j] = up_b * sum_c inx_w + inx_b
__global__ void k_wsum(const float* inxw, const float* inxb, const float* upb,
                       float* wsum) {
  int i = blockIdx.x * blockDim.x + threadIdx.x;
  if (i >= 9 * 384) return;
  float s = 0.f;
  const float* wp = inxw + i * 486;
  for (int c = 0; c < 486; ++c) s += wp[c];
  wsum[i] = upb[0] * s + inxb[i];
}

// ---------------------------------------------------------------- repacks
__global__ void k_cvt(const float* s, _Float16* d, int n) {
  int i = blockIdx.x * blockDim.x + threadIdx.x;
  if (i < n) d[i] = (_Float16)s[i];
}

// dilh_w (9,384,192,6) -> wd16 [l][k][j][c]
__global__ void k_cvt_dilw(const float* s, _Float16* d) {
  int i = blockIdx.x * blockDim.x + threadIdx.x;
  if (i >= 9 * 6 * 384 * 192) return;
  int c = i % 192, j = (i / 192) % 384, k = (i / (192 * 384)) % 6,
      l = i / (192 * 384 * 6);
  d[i] = (_Float16)s[((l * 384 + j) * 192 + c) * 6 + k];
}

// causal_w (192,256,6) -> wc16 [k][j][c]
__global__ void k_cvt_causw(const float* s, _Float16* d) {
  int i = blockIdx.x * blockDim.x + threadIdx.x;
  if (i >= 6 * 192 * 256) return;
  int c = i % 256, j = (i / 256) % 192, k = i / (256 * 192);
  d[i] = (_Float16)s[(j * 256 + c) * 6 + k];
}

// audio (B,256,T) f32 -> padded f16 (B,256,ASTR), 5 zeros left, 16 right
__global__ void k_fill_audio(const float* s, _Float16* d) {
  int i = blockIdx.x * blockDim.x + threadIdx.x;
  if (i >= B_ * NQ * ASTR) return;
  int t = i % ASTR - 5;
  int c = (i / ASTR) % NQ, b = i / (ASTR * NQ);
  float v = (t >= 0 && t < T_) ? s[(b * NQ + c) * T_ + t] : 0.f;
  d[i] = (_Float16)v;
}

// ---------------------------------------------------------------- causal conv
__global__ void __launch_bounds__(BDIM)
k_causal(const _Float16* au16, const _Float16* wc16, const float* cb,
         _Float16* hout) {
  __shared__ _Float16 sA[6][16][256];  // 49152 B
  int b = blockIdx.x / NT16, t0 = (blockIdx.x % NT16) * 16;
  int tid = threadIdx.x, lane = tid & 31, wave = tid >> 5, hi = lane >> 4;

  for (int i = tid; i < 256 * 96; i += BDIM) {  // i = c*96 + k*16 + n
    int c = i / 96, k = (i % 96) / 16, n = i % 16;
    int t = t0 + n - (5 - k);
    sA[k][n][c] = au16[(b * NQ + c) * ASTR + 5 + t];
  }
  __syncthreads();

  for (int mt = wave; mt < 12; mt += 8) {
    int j0 = mt * 16;
    v8f acc;
#pragma unroll
    for (int r = 0; r < 8; ++r) acc[r] = cb[j0 + hi * 8 + r];
#pragma unroll
    for (int k = 0; k < 6; ++k)
      for (int c0 = 0; c0 < 256; c0 += 32) {
        v16h a = loadA(wc16 + (k * HID + j0) * 256 + c0, 256, lane);
        v16h bb = loadB(&sA[k][0][0] + c0, 256, lane);
        acc = wmma16(a, bb, acc);
      }
#pragma unroll
    for (int r = 0; r < 8; ++r) {
      int m = hi * 8 + r, n = lane & 15, t = t0 + n;
      if (t < T_) {
        float x = acc[r];
        hout[(b * HID + j0 + m) * HSTR + PADH + t] =
            (_Float16)(x / (1.f + fabsf(x)));
      }
    }
  }
}

// ---------------------------------------------------------------- WaveNet layer
__global__ void __launch_bounds__(BDIM)
k_layer(int l, int dil, const _Float16* hin, _Float16* hout,
        const _Float16* wd16, const float* dilb, const float* g,
        const float* wsum, const float* upw, const _Float16* ws16,
        const float* skb, float* ssum) {
  __shared__ _Float16 sH[6][16][192];   // 36864 B (tap windows, [k][n][c])
  __shared__ float sXH[384][16];        // 24576 B
  _Float16* sHN = &sH[0][0][0];         // reused as h_new [n][c] after gating

  int b = blockIdx.x / NT16, t0 = (blockIdx.x % NT16) * 16;
  int tid = threadIdx.x, lane = tid & 31, wave = tid >> 5, hi = lane >> 4;

  for (int i = tid; i < 192 * 96; i += BDIM) {  // i = c*96 + k*16 + n
    int c = i / 96, k = (i % 96) / 16, n = i % 16;
    int t = t0 + n - (5 - k) * dil;
    sH[k][n][c] = hin[(b * HID + c) * HSTR + PADH + t];
  }
  __syncthreads();

  // gate pre-activations: dil-conv GEMM, then multiply by folded conditioning
#pragma unroll
  for (int it = 0; it < 3; ++it) {
    int j0 = (wave * 3 + it) * 16;
    v8f acc;
#pragma unroll
    for (int r = 0; r < 8; ++r) acc[r] = dilb[l * 384 + j0 + hi * 8 + r];
#pragma unroll
    for (int k = 0; k < 6; ++k)
      for (int c0 = 0; c0 < 192; c0 += 32) {
        v16h a = loadA(wd16 + ((l * 6 + k) * 384 + j0) * 192 + c0, 192, lane);
        v16h bb = loadB(&sH[k][0][0] + c0, 192, lane);
        acc = wmma16(a, bb, acc);
      }
#pragma unroll
    for (int r = 0; r < 8; ++r) {
      int m = hi * 8 + r, n = lane & 15, j = j0 + m;
      int t = t0 + n, te = t < T_ ? t : T_ - 1;
      int q = (te + 1) / UPF, rr = (te + 1) - q * UPF;
      float inx = upw[rr] * g[((l * B_ + b) * 384 + j) * TAUX + q] +
                  wsum[l * 384 + j];
      sXH[j][n] = acc[r] * inx;
    }
  }
  __syncthreads();

  // gating: z = sigmoid(xh[:H]); h = (1-z)*tanh(xh[H:]) + z*h_old
  for (int i = tid; i < HID * 16; i += BDIM) {
    int c = i / 16, n = i % 16;
    float zx = sXH[c][n], hx = sXH[c + HID][n];
    float hold = (float)sH[5][n][c];  // tap k=5 is offset 0 -> h at t0+n
    float z = 1.f / (1.f + expf(-zx));
    float hn = (1.f - z) * tanhf(hx) + z * hold;
    sHN[n * 192 + c] = (_Float16)hn;  // writes sH[0] region, disjoint from sH[5]
    int t = t0 + n;
    if (t < T_) hout[(b * HID + c) * HSTR + PADH + t] = (_Float16)hn;
  }
  __syncthreads();

  // skip GEMM, accumulate into global sum
#pragma unroll
  for (int it = 0; it < 2; ++it) {
    int o0 = (wave * 2 + it) * 16;
    v8f acc;
#pragma unroll
    for (int r = 0; r < 8; ++r) acc[r] = skb[l * SKIPC + o0 + hi * 8 + r];
#pragma unroll
    for (int c0 = 0; c0 < 192; c0 += 32) {
      v16h a = loadA(ws16 + (l * SKIPC + o0) * 192 + c0, 192, lane);
      v16h bb = loadB(sHN + c0, 192, lane);
      acc = wmma16(a, bb, acc);
    }
#pragma unroll
    for (int r = 0; r < 8; ++r) {
      int m = hi * 8 + r, n = lane & 15, t = t0 + n;
      if (t < T_) {
        float* p = &ssum[(b * SKIPC + o0 + m) * T_ + t];
        *p = (l == 0) ? acc[r] : (*p + acc[r]);
      }
    }
  }
}

// ---------------------------------------------------------------- output head
__global__ void __launch_bounds__(BDIM)
k_out(const float* ssum, const _Float16* w1, const float* b1,
      const _Float16* w2, const float* b2, float* out) {
  __shared__ _Float16 sB1[16][256];
  __shared__ _Float16 sB2[16][256];
  int b = blockIdx.x / NT16, t0 = (blockIdx.x % NT16) * 16;
  int tid = threadIdx.x, lane = tid & 31, wave = tid >> 5, hi = lane >> 4;

  for (int i = tid; i < 16 * 256; i += BDIM) {
    int n = i / 256, c = i % 256, t = t0 + n;
    float v = (t < T_) ? fmaxf(ssum[(b * SKIPC + c) * T_ + t], 0.f) : 0.f;
    sB1[n][c] = (_Float16)v;
  }
  __syncthreads();

#pragma unroll
  for (int it = 0; it < 2; ++it) {
    int o0 = (wave * 2 + it) * 16;
    v8f acc;
#pragma unroll
    for (int r = 0; r < 8; ++r) acc[r] = b1[o0 + hi * 8 + r];
#pragma unroll
    for (int c0 = 0; c0 < 256; c0 += 32) {
      v16h a = loadA(w1 + o0 * 256 + c0, 256, lane);
      v16h bb = loadB(&sB1[0][0] + c0, 256, lane);
      acc = wmma16(a, bb, acc);
    }
#pragma unroll
    for (int r = 0; r < 8; ++r) {
      int m = hi * 8 + r, n = lane & 15;
      sB2[n][o0 + m] = (_Float16)fmaxf(acc[r], 0.f);
    }
  }
  __syncthreads();

#pragma unroll
  for (int it = 0; it < 2; ++it) {
    int o0 = (wave * 2 + it) * 16;
    v8f acc;
#pragma unroll
    for (int r = 0; r < 8; ++r) acc[r] = b2[o0 + hi * 8 + r];
#pragma unroll
    for (int c0 = 0; c0 < 256; c0 += 32) {
      v16h a = loadA(w2 + o0 * 256 + c0, 256, lane);
      v16h bb = loadB(&sB2[0][0] + c0, 256, lane);
      acc = wmma16(a, bb, acc);
    }
#pragma unroll
    for (int r = 0; r < 8; ++r) {
      int m = hi * 8 + r, n = lane & 15, t = t0 + n;
      if (t < T_) out[(b * T_ + t) * NQ + o0 + m] = acc[r];
    }
  }
}

// ---------------------------------------------------------------- launcher
static inline int cdiv(long long a, int b) { return (int)((a + b - 1) / b); }

extern "C" void kernel_launch(void* const* d_in, const int* in_sizes, int n_in,
                              void* d_out, int out_size, void* d_ws,
                              size_t ws_size, hipStream_t stream) {
  const float* audio = (const float*)d_in[0];
  const float* aux   = (const float*)d_in[1];
  const float* siw   = (const float*)d_in[2];
  const float* sib   = (const float*)d_in[3];
  const float* a0w   = (const float*)d_in[4];
  const float* a0b   = (const float*)d_in[5];
  const float* a1w   = (const float*)d_in[6];
  const float* a1b   = (const float*)d_in[7];
  const float* upw   = (const float*)d_in[8];
  const float* upb   = (const float*)d_in[9];
  const float* cw    = (const float*)d_in[10];
  const float* cb    = (const float*)d_in[11];
  const float* inxw  = (const float*)d_in[12];
  const float* inxb  = (const float*)d_in[13];
  const float* dilw  = (const float*)d_in[14];
  const float* dilb  = (const float*)d_in[15];
  const float* skw   = (const float*)d_in[16];
  const float* skb   = (const float*)d_in[17];
  const float* o1w   = (const float*)d_in[18];
  const float* o1b   = (const float*)d_in[19];
  const float* o2w   = (const float*)d_in[20];
  const float* o2b   = (const float*)d_in[21];

  char* base = (char*)d_ws;
  size_t off = 0;
  auto alloc = [&](size_t bytes) -> char* {
    off = (off + 255) & ~(size_t)255;
    char* p = base + off;
    off += bytes;
    return p;
  };
  float* a0   = (float*)alloc((size_t)B_ * NAUX * TAUX * 4);
  float* a1   = (float*)alloc((size_t)B_ * 162 * TAUX * 4);
  float* a2   = (float*)alloc((size_t)B_ * 486 * TAUX * 4);
  float* g    = (float*)alloc((size_t)9 * B_ * 384 * TAUX * 4);
  float* wsum = (float*)alloc((size_t)9 * 384 * 4);
  _Float16* wd16 = (_Float16*)alloc((size_t)9 * 6 * 384 * 192 * 2);
  _Float16* ws16 = (_Float16*)alloc((size_t)9 * 256 * 192 * 2);
  _Float16* wc16 = (_Float16*)alloc((size_t)6 * 192 * 256 * 2);
  _Float16* w116 = (_Float16*)alloc((size_t)256 * 256 * 2);
  _Float16* w216 = (_Float16*)alloc((size_t)256 * 256 * 2);
  _Float16* au16 = (_Float16*)alloc((size_t)B_ * NQ * ASTR * 2);
  size_t hbytes  = (size_t)B_ * HID * HSTR * 2;
  _Float16* hb0  = (_Float16*)alloc(hbytes);
  _Float16* hb1  = (_Float16*)alloc(hbytes);
  float* ssum    = (float*)alloc((size_t)B_ * SKIPC * T_ * 4);

  // zero h buffers (pads must be zero; pad region is never written by kernels)
  hipMemsetAsync(hb0, 0, (size_t)((char*)ssum - (char*)hb0), stream);

  k_aux_scale<<<cdiv(B_ * NAUX * TAUX, BDIM), BDIM, 0, stream>>>(aux, siw, sib, a0);
  k_aux0<<<cdiv(B_ * 162 * TAUX, BDIM), BDIM, 0, stream>>>(a0, a0w, a0b, a1);
  k_aux1<<<cdiv(B_ * 486 * TAUX, BDIM), BDIM, 0, stream>>>(a1, a1w, a1b, a2);
  k_cond<<<cdiv((long long)9 * B_ * 384 * TAUX, BDIM), BDIM, 0, stream>>>(a2, inxw, g);
  k_wsum<<<cdiv(9 * 384, BDIM), BDIM, 0, stream>>>(inxw, inxb, upb, wsum);

  k_cvt_dilw<<<cdiv((long long)9 * 6 * 384 * 192, BDIM), BDIM, 0, stream>>>(dilw, wd16);
  k_cvt<<<cdiv(9 * 256 * 192, BDIM), BDIM, 0, stream>>>(skw, ws16, 9 * 256 * 192);
  k_cvt_causw<<<cdiv(6 * 192 * 256, BDIM), BDIM, 0, stream>>>(cw, wc16);
  k_cvt<<<cdiv(256 * 256, BDIM), BDIM, 0, stream>>>(o1w, w116, 256 * 256);
  k_cvt<<<cdiv(256 * 256, BDIM), BDIM, 0, stream>>>(o2w, w216, 256 * 256);
  k_fill_audio<<<cdiv((long long)B_ * NQ * ASTR, BDIM), BDIM, 0, stream>>>(audio, au16);

  dim3 gridT(B_ * NT16);
  k_causal<<<gridT, BDIM, 0, stream>>>(au16, wc16, cb, hb0);

  const int dilf[9] = {1, 6, 36, 1, 6, 36, 1, 6, 36};
  _Float16* hin = hb0;
  _Float16* hout = hb1;
  for (int l = 0; l < 9; ++l) {
    k_layer<<<gridT, BDIM, 0, stream>>>(l, dilf[l], hin, hout, wd16, dilb, g,
                                        wsum, upw, ws16, skb, ssum);
    _Float16* tmp = hin; hin = hout; hout = tmp;
  }

  k_out<<<gridT, BDIM, 0, stream>>>(ssum, w116, o1b, w216, o2b, (float*)d_out);
}
